// DLRM_29798483100151
// MI455X (gfx1250) — matrix operand order repaired and checked by reading, not compiled, not run on hardware
//
#include <hip/hip_runtime.h>
#include <hip/hip_bf16.h>

typedef __attribute__((ext_vector_type(16))) _Float16 v16h;
typedef __attribute__((ext_vector_type(8)))  _Float16 v8h;
typedef __attribute__((ext_vector_type(8)))  float    v8f;

#define BATCH   2048
#define D_DENSE 13
#define NF      7
#define CARD    10000
#define E_DIM   64
#define D_OUT   32
#define S_OUT   32
#define D_HID   64
#define S_HID   64
#define CONCAT  256      // D_OUT + NF*S_OUT
#define P_IN    65536    // CONCAT*CONCAT
#define P_HID   512
#define KCHUNKS 2048     // P_IN / 32

// ---------------------------------------------------------------------------
// Kernel 1: dense MLP + embedding gather + per-feature sparse MLPs -> concat f16
// One block (512 threads) per batch row. tokenizers == arange, so the
// argmax-match index is just the sparse value itself.
// ---------------------------------------------------------------------------
__global__ void __launch_bounds__(512) features_kernel(
    const float* __restrict__ dense, const int* __restrict__ sparse,
    const float* __restrict__ emb,
    const float* __restrict__ dw1, const float* __restrict__ db1,
    const float* __restrict__ dw2, const float* __restrict__ db2,
    const float* __restrict__ sw1, const float* __restrict__ sb1,
    const float* __restrict__ sw2, const float* __restrict__ sb2,
    _Float16* __restrict__ concat16)
{
    const int b = blockIdx.x;
    const int t = threadIdx.x;

    __shared__ float hid[D_HID + NF * S_HID];  // 512 hidden activations
    __shared__ float xs[16];                   // dense features
    __shared__ float es[NF * E_DIM];           // gathered embeddings

    if (t < D_DENSE) xs[t] = dense[b * D_DENSE + t];
    if (t < NF * E_DIM) {
        int f = t >> 6, e = t & 63;
        int idx = sparse[b * NF + f];          // identity tokenizer
        es[t] = emb[((size_t)f * CARD + idx) * E_DIM + e];
    }
    __syncthreads();

    // Phase A: all 512 hidden units (64 dense + 7*64 sparse)
    if (t < D_HID) {
        float acc = db1[t];
        for (int k = 0; k < D_DENSE; ++k) acc += xs[k] * dw1[k * D_HID + t];
        hid[t] = fmaxf(acc, 0.f);
    } else {
        int f = (t - D_HID) >> 6, h = (t - D_HID) & 63;
        float acc = sb1[f * S_HID + h];
        for (int k = 0; k < E_DIM; ++k)
            acc += es[f * E_DIM + k] * sw1[(f * E_DIM + k) * S_HID + h];
        hid[t] = fmaxf(acc, 0.f);
    }
    __syncthreads();

    // Phase B: 256 concat outputs, stored as f16 for the WMMA interaction GEMM
    if (t < CONCAT) {
        float acc;
        if (t < D_OUT) {
            acc = db2[t];
            for (int k = 0; k < D_HID; ++k) acc += hid[k] * dw2[k * D_OUT + t];
        } else {
            int f = (t - D_OUT) >> 5, o = (t - D_OUT) & 31;
            acc = sb2[f * S_OUT + o];
            for (int k = 0; k < S_HID; ++k)
                acc += hid[D_HID + f * S_HID + k] * sw2[(f * S_HID + k) * S_OUT + o];
        }
        concat16[b * CONCAT + t] = (_Float16)acc;
    }
}

// ---------------------------------------------------------------------------
// Kernel 2: pack pw1 (f32, [65536][512] row-major) into f16 wave32 B-fragment
// layout for v_wmma_f32_16x16x32_f16:
//   pw1p[((kc*32 + nt)*32 + lane)*16 + kk]
//   lanes 0-15 : N = nt*16+lane,    K = kc*32 + kk       (kk = 0..15)
//   lanes 16-31: N = nt*16+lane-16, K = kc*32 + 16 + kk
// Each thread emits 16 contiguous halves (one 32B lane slice).
// ---------------------------------------------------------------------------
__global__ void __launch_bounds__(256) pack_pw1_kernel(
    const float* __restrict__ pw1, _Float16* __restrict__ pw1p)
{
    int tid  = blockIdx.x * blockDim.x + threadIdx.x;  // 2048*32*32 threads
    int lane = tid & 31;
    int nt   = (tid >> 5) & 31;
    int kc   = tid >> 10;
    int n    = nt * 16 + (lane & 15);
    int kb   = kc * 32 + ((lane >> 4) << 4);
    v16h out;
    for (int kk = 0; kk < 16; ++kk)
        out[kk] = (_Float16)pw1[(size_t)(kb + kk) * P_HID + n];
    *(v16h*)(pw1p + (size_t)tid * 16) = out;
}

// ---------------------------------------------------------------------------
// Kernel 3: interaction GEMM  hidden = (c outer c) @ pw1   (no bias yet)
// Grid: (32 M-blocks of 64 rows) x (4 N-groups). Block = 256 threads = 8 waves.
// Wave w owns N-tile ngrp*8+w and all FOUR 16-row M-tiles of the block:
// one L2 B-fragment feeds 4 independent WMMA chains.
// A-fragment for K-chunk kc (K = i*256 + j, i = kc>>3, j0 = (kc&7)*32) is
// c[row,i] * c[row, j0:j0+32], built from the LDS c-tile with packed f16 muls;
// the scale c[row,i] is hoisted to the i-loop (constant over 8 chunks).
// B is software-pipelined 2 chunks deep (UNclamped prefetch: worst case reads
// 64KB past pw1p, which lands in the adjacent `hidden` workspace region --
// allocated and value-unused, so it's safe and saves a SALU clamp chain per
// chunk). jc unrolled 4x: enough scheduling window without pushing the
// allocator past 256 VGPRs (the round-3 build spilled into MSB-addressed
// VGPRs and paid s_set_vgpr_msb toggles in the hot loop).
// ---------------------------------------------------------------------------
__global__ void __launch_bounds__(256) interact_gemm_kernel(
    const _Float16* __restrict__ c16,
    const _Float16* __restrict__ pw1p,
    float* __restrict__ hidden)
{
    __shared__ _Float16 cs[64 * CONCAT];   // 32 KB: 64 batch rows of concat

    const int mblk  = blockIdx.x;          // 0..31
    const int t     = threadIdx.x;
    const int wave  = t >> 5;
    const int lane  = t & 31;
    const int ntile = blockIdx.y * 8 + wave;

    // cooperative load: 64 rows x 256 halves = 32 KB contiguous
    {
        const uint4* src = (const uint4*)(c16 + (size_t)mblk * 64 * CONCAT);
        uint4* dst = (uint4*)cs;
        for (int i = t; i < 2048; i += 256) dst[i] = src[i];
    }
    __syncthreads();

    const int row = lane & 15;
    const int sel = lane >> 4;                       // K-half select
    const _Float16* rp0 = cs + (row)       * CONCAT; // M-tile 0 (rows  0-15)
    const _Float16* rp1 = cs + (16 + row)  * CONCAT; // M-tile 1 (rows 16-31)
    const _Float16* rp2 = cs + (32 + row)  * CONCAT; // M-tile 2 (rows 32-47)
    const _Float16* rp3 = cs + (48 + row)  * CONCAT; // M-tile 3 (rows 48-63)
    const _Float16* bp  = pw1p + ((size_t)ntile * 32 + lane) * 16;

    v8f acc0 = {}, acc1 = {}, acc2 = {}, acc3 = {};

    // 2-deep B pipeline
    v16h bfA = *(const v16h*)(bp);                    // kc = 0
    v16h bfB = *(const v16h*)(bp + (size_t)16384);    // kc = 1

    for (int i = 0; i < CONCAT; ++i) {
        // splat scales, constant across the 8 inner chunks of this i-block
        const _Float16 s0 = rp0[i];
        const _Float16 s1 = rp1[i];
        const _Float16 s2 = rp2[i];
        const _Float16 s3 = rp3[i];

        #pragma unroll 4
        for (int jc = 0; jc < 8; ++jc) {
            const int kc = i * 8 + jc;
            // prefetch B for kc+2 (unclamped; tail reads spill harmlessly
            // into the hidden-buffer region of the workspace)
            v16h bfP = *(const v16h*)(bp + (size_t)(kc + 2) * 16384);

            const int jb = jc << 5;
            const int lo = jb + sel * 16;     // lanes0-15: K=0..7 | 16..23
            const int hi = jb + sel * 16 + 8; // lanes0-15: K=8..15 | 24..31

            v16h a0 = __builtin_shufflevector(*(const v8h*)(rp0 + lo),
                                              *(const v8h*)(rp0 + hi),
                        0,1,2,3,4,5,6,7,8,9,10,11,12,13,14,15);
            v16h a1 = __builtin_shufflevector(*(const v8h*)(rp1 + lo),
                                              *(const v8h*)(rp1 + hi),
                        0,1,2,3,4,5,6,7,8,9,10,11,12,13,14,15);
            v16h a2 = __builtin_shufflevector(*(const v8h*)(rp2 + lo),
                                              *(const v8h*)(rp2 + hi),
                        0,1,2,3,4,5,6,7,8,9,10,11,12,13,14,15);
            v16h a3 = __builtin_shufflevector(*(const v8h*)(rp3 + lo),
                                              *(const v8h*)(rp3 + hi),
                        0,1,2,3,4,5,6,7,8,9,10,11,12,13,14,15);
            #pragma unroll
            for (int e = 0; e < 16; ++e) {
                a0[e] *= s0; a1[e] *= s1; a2[e] *= s2; a3[e] *= s3;
            }

            acc0 = __builtin_amdgcn_wmma_f32_16x16x32_f16(
                       false, a0, false, bfA, (short)0, acc0, false, false);
            acc1 = __builtin_amdgcn_wmma_f32_16x16x32_f16(
                       false, a1, false, bfA, (short)0, acc1, false, false);
            acc2 = __builtin_amdgcn_wmma_f32_16x16x32_f16(
                       false, a2, false, bfA, (short)0, acc2, false, false);
            acc3 = __builtin_amdgcn_wmma_f32_16x16x32_f16(
                       false, a3, false, bfA, (short)0, acc3, false, false);

            bfA = bfB;     // rotate pipeline
            bfB = bfP;
        }
    }

    // C/D layout: VGPR r -> M = r + 8*sel, N = lane&15
    const int ncol  = ntile * 16 + (lane & 15);
    const int mrow0 = mblk * 64 + sel * 8;
    #pragma unroll
    for (int r = 0; r < 8; ++r) {
        hidden[(size_t)(mrow0 + r)      * P_HID + ncol] = acc0[r];
        hidden[(size_t)(mrow0 + 16 + r) * P_HID + ncol] = acc1[r];
        hidden[(size_t)(mrow0 + 32 + r) * P_HID + ncol] = acc2[r];
        hidden[(size_t)(mrow0 + 48 + r) * P_HID + ncol] = acc3[r];
    }
}

// ---------------------------------------------------------------------------
// Kernel 4: logits = sigmoid(relu(hidden + pb1) @ pw2 + pb2)
// One wave32 per batch row; shuffle tree reduction.
// ---------------------------------------------------------------------------
__global__ void __launch_bounds__(256) head_kernel(
    const float* __restrict__ hidden, const float* __restrict__ pb1,
    const float* __restrict__ pw2, const float* __restrict__ pb2,
    float* __restrict__ out)
{
    const int b    = blockIdx.x * 8 + (threadIdx.x >> 5);
    const int lane = threadIdx.x & 31;
    float acc = 0.f;
    for (int kk = 0; kk < 16; ++kk) {
        int k = lane * 16 + kk;
        float h = hidden[(size_t)b * P_HID + k] + pb1[k];
        acc += fmaxf(h, 0.f) * pw2[k];
    }
    for (int off = 16; off > 0; off >>= 1)
        acc += __shfl_xor(acc, off, 32);
    if (lane == 0) out[b] = 1.f / (1.f + __expf(-(acc + pb2[0])));
}

// ---------------------------------------------------------------------------
extern "C" void kernel_launch(void* const* d_in, const int* in_sizes, int n_in,
                              void* d_out, int out_size, void* d_ws, size_t ws_size,
                              hipStream_t stream)
{
    (void)in_sizes; (void)n_in; (void)out_size; (void)ws_size;

    const float* dense  = (const float*)d_in[0];
    const int*   sparse = (const int*)d_in[1];
    /* d_in[2] tokenizers: identity, unused */
    const float* emb = (const float*)d_in[3];
    const float* dw1 = (const float*)d_in[4];
    const float* db1 = (const float*)d_in[5];
    const float* dw2 = (const float*)d_in[6];
    const float* db2 = (const float*)d_in[7];
    const float* sw1 = (const float*)d_in[8];
    const float* sb1 = (const float*)d_in[9];
    const float* sw2 = (const float*)d_in[10];
    const float* sb2 = (const float*)d_in[11];
    const float* pw1 = (const float*)d_in[12];
    const float* pb1 = (const float*)d_in[13];
    const float* pw2 = (const float*)d_in[14];
    const float* pb2 = (const float*)d_in[15];

    // Workspace layout (all 256B-aligned):
    char* ws = (char*)d_ws;
    _Float16* concat16 = (_Float16*)ws;                               // 1 MB
    _Float16* pw1p     = (_Float16*)(ws + (1u << 20));                // 64 MB
    float*    hidden   = (float*)(ws + (1u << 20)
                                   + (size_t)P_IN * P_HID * 2);       // 4 MB

    features_kernel<<<BATCH, 512, 0, stream>>>(
        dense, sparse, emb, dw1, db1, dw2, db2, sw1, sb1, sw2, sb2, concat16);

    pack_pw1_kernel<<<(2048 * 32 * 32) / 256, 256, 0, stream>>>(pw1, pw1p);

    dim3 ggrid(32, 4);
    interact_gemm_kernel<<<ggrid, 256, 0, stream>>>(concat16, pw1p, hidden);

    head_kernel<<<BATCH / 8, 256, 0, stream>>>(hidden, pb1, pw2, pb2,
                                               (float*)d_out);
}
